// SupConLoss_16621523436342
// MI455X (gfx1250) — matrix operand order repaired and checked by reading, not compiled, not run on hardware
//
#include <hip/hip_runtime.h>

// SupCon-style loss for N=4096, D=768, C=10 on gfx1250 (MI455X).
// f16 WMMA Gram sweeps (v_wmma_f32_16x16x32_f16) with fused, BRANCHLESS per-row
// statistic epilogues; no 67MB Gram materialization; no atomics (deterministic).
// Workspace need: ~6.4 MB (Nh f16 4096x768 + 5 float[4096] stat arrays).

#define NROWS  4096
#define DIMK   768
#define KTILES (DIMK / 32)    // 24
#define CTILES (NROWS / 16)   // 256
#define NWAVES 8
#define NCLASS 10
#define NEGBIG (-3.0e38f)

typedef __attribute__((ext_vector_type(16))) _Float16 v16h;
typedef __attribute__((ext_vector_type(8)))  _Float16 v8h;
typedef __attribute__((ext_vector_type(8)))  float    v8f;

// ---------------- normalize rows + convert to f16 ----------------
__global__ __launch_bounds__(256) void normalize_kernel(const float* __restrict__ X,
                                                        _Float16* __restrict__ Nh) {
  const int row = blockIdx.x;
  const int t   = threadIdx.x;
  const float* xr = X + (size_t)row * DIMK;
  float v0 = xr[t], v1 = xr[t + 256], v2 = xr[t + 512];
  float s = v0 * v0 + v1 * v1 + v2 * v2;
  __shared__ float red[256];
  red[t] = s; __syncthreads();
  for (int o = 128; o > 0; o >>= 1) { if (t < o) red[t] += red[t + o]; __syncthreads(); }
  float inv = 1.0f / fmaxf(sqrtf(red[0]), 1e-12f);
  _Float16* out = Nh + (size_t)row * DIMK;
  out[t]       = (_Float16)(v0 * inv);
  out[t + 256] = (_Float16)(v1 * inv);
  out[t + 512] = (_Float16)(v2 * inv);
}

// ---------------- WMMA fragment loads (row-major f16, 16B aligned) ----------------
// A 16x32 f16: lane<16 holds M=lane, K{0..7,16..23}; lane>=16 holds K{8..15,24..31}.
__device__ __forceinline__ v16h load_a_frag(const _Float16* __restrict__ rowp, int k0, int h) {
  v8h lo = *(const v8h*)(rowp + k0 + 8 * h);
  v8h hi = *(const v8h*)(rowp + k0 + 16 + 8 * h);
  return __builtin_shufflevector(lo, hi, 0,1,2,3,4,5,6,7,8,9,10,11,12,13,14,15);
}
// B 32x16 f16: lane<16 holds N=lane, K{0..7,16..23}; lane>=16 holds K{8..15,24..31}.
__device__ __forceinline__ v16h load_b_frag(const _Float16* __restrict__ rowp, int k0, int h) {
  v8h lo = *(const v8h*)(rowp + k0 + 8 * h);
  v8h hi = *(const v8h*)(rowp + k0 + 16 + 8 * h);
  return __builtin_shufflevector(lo, hi, 0,1,2,3,4,5,6,7,8,9,10,11,12,13,14,15);
}

// 16x16 G tile for (mi, nj), K=768, dual accumulators to relax the C->D chain.
__device__ __forceinline__ void gram_tile(const _Float16* __restrict__ Arow,
                                          const _Float16* __restrict__ Brow,
                                          int h, v8f& acc0, v8f& acc1) {
  #pragma unroll 4
  for (int kt = 0; kt < KTILES; kt += 2) {
    const int k0 = kt * 32;
    v16h a0 = load_a_frag(Arow, k0, h);
    v16h b0 = load_b_frag(Brow, k0, h);
    acc0 = __builtin_amdgcn_wmma_f32_16x16x32_f16(false, a0, false, b0, (short)0, acc0,
                                                  false, false);
    v16h a1 = load_a_frag(Arow, k0 + 32, h);
    v16h b1 = load_b_frag(Brow, k0 + 32, h);
    acc1 = __builtin_amdgcn_wmma_f32_16x16x32_f16(false, a1, false, b1, (short)0, acc1,
                                                  false, false);
  }
}

// ---------------- pass A: rowmax(offdiag), rowmax(neg), sum over positives ----------------
__global__ __launch_bounds__(32 * NWAVES) void gram_pass_a(const _Float16* __restrict__ Nh,
                                                           const int* __restrict__ labels,
                                                           float* __restrict__ rowmax_off,
                                                           float* __restrict__ rowmax_neg,
                                                           float* __restrict__ possum) {
  const int mi   = blockIdx.x;          // row tile (16 rows)
  const int tid  = threadIdx.x;
  const int wid  = tid >> 5;
  const int lane = tid & 31;
  const int h    = lane >> 4;
  const int l15  = lane & 15;

  int rowLab[8];
  #pragma unroll
  for (int v = 0; v < 8; ++v) rowLab[v] = labels[mi * 16 + 8 * h + v];

  float mxo[8], mxn[8], ps[8];
  #pragma unroll
  for (int v = 0; v < 8; ++v) { mxo[v] = NEGBIG; mxn[v] = NEGBIG; ps[v] = 0.0f; }

  const _Float16* Arow = Nh + (size_t)(mi * 16 + l15) * DIMK;

  for (int nj = wid; nj < CTILES; nj += NWAVES) {
    const _Float16* Brow = Nh + (size_t)(nj * 16 + l15) * DIMK;
    v8f acc0 = {}, acc1 = {};
    gram_tile(Arow, Brow, h, acc0, acc1);
    const int col  = nj * 16 + l15;
    const int labJ = labels[col];
    #pragma unroll
    for (int v = 0; v < 8; ++v) {
      const int   row  = mi * 16 + v + 8 * h;
      const float g    = acc0[v] + acc1[v];
      const bool  offd = (row != col);
      const bool  neg  = (labJ != rowLab[v]);
      mxo[v] = fmaxf(mxo[v], offd ? g : NEGBIG);            // v_cndmask + v_max
      mxn[v] = fmaxf(mxn[v], neg ? g : NEGBIG);
      ps[v] += (offd && !neg) ? g : 0.0f;                   // positives: same label, j != i
    }
  }

  // reduce across the 16 lanes sharing each row (offsets stay inside 16-lane groups)
  #pragma unroll
  for (int v = 0; v < 8; ++v) {
    #pragma unroll
    for (int o = 8; o >= 1; o >>= 1) {
      mxo[v] = fmaxf(mxo[v], __shfl_xor(mxo[v], o));
      mxn[v] = fmaxf(mxn[v], __shfl_xor(mxn[v], o));
      ps[v] += __shfl_xor(ps[v], o);
    }
  }

  __shared__ float s_o[NWAVES][16], s_n[NWAVES][16], s_p[NWAVES][16];
  if (l15 == 0) {
    #pragma unroll
    for (int v = 0; v < 8; ++v) {
      s_o[wid][8 * h + v] = mxo[v];
      s_n[wid][8 * h + v] = mxn[v];
      s_p[wid][8 * h + v] = ps[v];
    }
  }
  __syncthreads();
  if (tid < 16) {
    float a = NEGBIG, b = NEGBIG, c = 0.0f;
    for (int w = 0; w < NWAVES; ++w) {
      a = fmaxf(a, s_o[w][tid]);
      b = fmaxf(b, s_n[w][tid]);
      c += s_p[w][tid];
    }
    rowmax_off[mi * 16 + tid] = a;
    rowmax_neg[mi * 16 + tid] = b;
    possum[mi * 16 + tid]     = c;
  }
}

// ---------------- pass B: denom1 = sum exp(10g-m1) (+diag), S2 = sum_neg exp(20g-m2) ----------------
__global__ __launch_bounds__(32 * NWAVES) void gram_pass_b(const _Float16* __restrict__ Nh,
                                                           const int* __restrict__ labels,
                                                           const float* __restrict__ rowmax_off,
                                                           const float* __restrict__ rowmax_neg,
                                                           float* __restrict__ denom1,
                                                           float* __restrict__ S2arr) {
  const int mi   = blockIdx.x;
  const int tid  = threadIdx.x;
  const int wid  = tid >> 5;
  const int lane = tid & 31;
  const int h    = lane >> 4;
  const int l15  = lane & 15;

  int rowLab[8]; float m1[8], m2[8];
  #pragma unroll
  for (int v = 0; v < 8; ++v) {
    const int row = mi * 16 + 8 * h + v;
    rowLab[v] = labels[row];
    m1[v] = fmaxf(10.0f * rowmax_off[row], 0.0f);   // diag of adt is zeroed -> max includes 0
    const float rn = rowmax_neg[row];
    m2[v] = (rn > -1.0e30f) ? 20.0f * rn : 0.0f;    // row_has_neg guard (matches reference)
  }

  float d1[8], s2[8];
  #pragma unroll
  for (int v = 0; v < 8; ++v) { d1[v] = 0.0f; s2[v] = 0.0f; }

  const _Float16* Arow = Nh + (size_t)(mi * 16 + l15) * DIMK;

  for (int nj = wid; nj < CTILES; nj += NWAVES) {
    const _Float16* Brow = Nh + (size_t)(nj * 16 + l15) * DIMK;
    v8f acc0 = {}, acc1 = {};
    gram_tile(Arow, Brow, h, acc0, acc1);
    const int col  = nj * 16 + l15;
    const int labJ = labels[col];
    #pragma unroll
    for (int v = 0; v < 8; ++v) {
      const int   row  = mi * 16 + v + 8 * h;
      const float g    = acc0[v] + acc1[v];
      const bool  offd = (row != col);
      const bool  neg  = (labJ != rowLab[v]);
      const float e1   = __expf(10.0f * g - m1[v]);         // v_exp_f32 (trans pipe)
      const float e2   = __expf(20.0f * g - m2[v]);
      d1[v] += offd ? e1 : 0.0f;                            // v_cndmask + v_add
      s2[v] += neg ? e2 : 0.0f;
    }
  }

  #pragma unroll
  for (int v = 0; v < 8; ++v) {
    #pragma unroll
    for (int o = 8; o >= 1; o >>= 1) {
      d1[v] += __shfl_xor(d1[v], o);
      s2[v] += __shfl_xor(s2[v], o);
    }
  }

  __shared__ float s_d[NWAVES][16], s_s[NWAVES][16];
  if (l15 == 0) {
    #pragma unroll
    for (int v = 0; v < 8; ++v) {
      s_d[wid][8 * h + v] = d1[v];
      s_s[wid][8 * h + v] = s2[v];
    }
  }
  __syncthreads();
  if (tid < 16) {
    const int row = mi * 16 + tid;
    float dd = 0.0f, ss = 0.0f;
    for (int w = 0; w < NWAVES; ++w) { dd += s_d[w][tid]; ss += s_s[w][tid]; }
    const float m1r = fmaxf(10.0f * rowmax_off[row], 0.0f);
    denom1[row] = dd + __expf(-m1r);   // diagonal of adt contributes exp(0 - m1)
    S2arr[row]  = ss;
  }
}

// ---------------- finalize: CE + per-row losses + scalar (single block, deterministic) ----------------
__global__ __launch_bounds__(256) void finalize_kernel(const float* __restrict__ predicts,
                                                       const int* __restrict__ labels,
                                                       const float* __restrict__ rowmax_off,
                                                       const float* __restrict__ possum,
                                                       const float* __restrict__ denom1,
                                                       const float* __restrict__ S2arr,
                                                       float* __restrict__ out) {
  const int t = threadIdx.x;
  __shared__ float shf[256];
  __shared__ int   shi[256];
  __shared__ float s_tot;
  __shared__ float s_pls[NCLASS];
  __shared__ int   s_cnt[NCLASS];

  int   cnt[NCLASS];
  float pls[NCLASS];
  for (int c = 0; c < NCLASS; ++c) { cnt[c] = 0; pls[c] = 0.0f; }
  float totS = 0.0f;
  for (int i = t; i < NROWS; i += 256) {
    int lab = labels[i];
    lab = lab < 0 ? 0 : (lab >= NCLASS ? NCLASS - 1 : lab);
    cnt[lab] += 1;
    const float s = S2arr[i];
    pls[lab] += s;
    totS += s;
  }
  shf[t] = totS; __syncthreads();
  for (int o = 128; o > 0; o >>= 1) { if (t < o) shf[t] += shf[t + o]; __syncthreads(); }
  if (t == 0) s_tot = shf[0];
  __syncthreads();
  for (int c = 0; c < NCLASS; ++c) {
    shi[t] = cnt[c]; shf[t] = pls[c]; __syncthreads();
    for (int o = 128; o > 0; o >>= 1) {
      if (t < o) { shi[t] += shi[t + o]; shf[t] += shf[t + o]; }
      __syncthreads();
    }
    if (t == 0) { s_cnt[c] = shi[0]; s_pls[c] = shf[0]; }
    __syncthreads();
  }
  const float totAll = s_tot;

  float ce = 0.0f, sum1 = 0.0f, sum2 = 0.0f;
  for (int i = t; i < NROWS; i += 256) {
    const float* pr = predicts + (size_t)i * NCLASS;
    float mx = pr[0];
    for (int c = 1; c < NCLASS; ++c) mx = fmaxf(mx, pr[c]);
    float se = 0.0f;
    for (int c = 0; c < NCLASS; ++c) se += __expf(pr[c] - mx);
    int lab = labels[i];
    lab = lab < 0 ? 0 : (lab >= NCLASS ? NCLASS - 1 : lab);
    ce += (mx + __logf(se)) - pr[lab];

    const int cn   = s_cnt[lab];
    const int msum = cn - 1;
    if (msum > 0) {
      const float m1 = fmaxf(10.0f * rowmax_off[i], 0.0f);
      sum1 += (10.0f * possum[i] - (float)msum * m1) / (float)msum
              - __logf(denom1[i] + 1e-12f);
      const int   negs = NROWS - cn;
      const float nds  = totAll - s_pls[lab];
      const float x    = (negs > 0) ? nds / (float)negs : 0.0f;
      sum2 += -__logf(x + 1e-12f);
    }
  }
  float ceT, s1T, s2T;
  shf[t] = ce; __syncthreads();
  for (int o = 128; o > 0; o >>= 1) { if (t < o) shf[t] += shf[t + o]; __syncthreads(); }
  ceT = shf[0]; __syncthreads();
  shf[t] = sum1; __syncthreads();
  for (int o = 128; o > 0; o >>= 1) { if (t < o) shf[t] += shf[t + o]; __syncthreads(); }
  s1T = shf[0]; __syncthreads();
  shf[t] = sum2; __syncthreads();
  for (int o = 128; o > 0; o >>= 1) { if (t < o) shf[t] += shf[t + o]; __syncthreads(); }
  s2T = shf[0];
  if (t == 0) {
    const float invN = 1.0f / (float)NROWS;
    // (1-ALPHA)*ce_mean + ALPHA*(-mean pos1) + 0.5*ALPHA*(-mean pos2), ALPHA=0.5
    out[0] = 0.5f * (ceT * invN) - 0.5f * (s1T * invN) - 0.25f * (s2T * invN);
  }
}

extern "C" void kernel_launch(void* const* d_in, const int* in_sizes, int n_in,
                              void* d_out, int out_size, void* d_ws, size_t ws_size,
                              hipStream_t stream) {
  const float* cls_feats = (const float*)d_in[0];
  const float* predicts  = (const float*)d_in[1];
  const int*   targets   = (const int*)d_in[2];

  char* ws = (char*)d_ws;
  size_t off = 0;
  _Float16* Nh = (_Float16*)(ws + off); off += (size_t)NROWS * DIMK * sizeof(_Float16); // 6,291,456
  float* rowmax_off = (float*)(ws + off); off += NROWS * sizeof(float);
  float* rowmax_neg = (float*)(ws + off); off += NROWS * sizeof(float);
  float* possum_a   = (float*)(ws + off); off += NROWS * sizeof(float);
  float* denom1     = (float*)(ws + off); off += NROWS * sizeof(float);
  float* S2arr      = (float*)(ws + off); off += NROWS * sizeof(float);

  normalize_kernel<<<NROWS, 256, 0, stream>>>(cls_feats, Nh);
  gram_pass_a<<<CTILES, 32 * NWAVES, 0, stream>>>(Nh, targets, rowmax_off, rowmax_neg,
                                                  possum_a);
  gram_pass_b<<<CTILES, 32 * NWAVES, 0, stream>>>(Nh, targets, rowmax_off, rowmax_neg,
                                                  denom1, S2arr);
  finalize_kernel<<<1, 256, 0, stream>>>(predicts, targets, rowmax_off, possum_a,
                                         denom1, S2arr, (float*)d_out);
}
